// Encoder_6347961663490
// MI455X (gfx1250) — matrix operand order repaired and checked by reading, not compiled, not run on hardware
//
#include <hip/hip_runtime.h>

// ---------------------------------------------------------------------------
// CDNA5 (gfx1250) implementation of the point-cloud encoder.
// GEMMs: v_wmma_f32_16x16x32_bf16 (fp32 accumulate), wave32.
//  - B macro-tile (32 cols x K) staged to LDS once per block, via
//    GLOBAL_LOAD_ASYNC_TO_LDS_B128 (ASYNCcnt) when available.
//  - each wave computes a 16x32 tile (2 WMMA accs share one A fragment)
//  - A fragment loads software-pipelined one k-step ahead.
// ---------------------------------------------------------------------------

typedef __attribute__((ext_vector_type(16))) __bf16 bf16x16;
typedef __attribute__((ext_vector_type(8)))  __bf16 bf16x8;
typedef __attribute__((ext_vector_type(8)))  float  f32x8;
typedef int v4i_gcc __attribute__((vector_size(16)));   // matches builtin param

#if defined(__HIP_DEVICE_COMPILE__) && \
    __has_builtin(__builtin_amdgcn_global_load_async_to_lds_b128)
#define ASYNC_LDS 1
#endif

// ------------------------- elementwise helpers ------------------------------
__global__ __launch_bounds__(256) void k_f32_to_bf16(const float* __restrict__ s,
                                                     __bf16* __restrict__ d, long n) {
  long i = (long)blockIdx.x * 256 + threadIdx.x;
  if (i < n) d[i] = (__bf16)s[i];
}

__global__ __launch_bounds__(256) void k_copy_f32(const float* __restrict__ s,
                                                  float* __restrict__ d, long n) {
  long i = (long)blockIdx.x * 256 + threadIdx.x;
  if (i < n) d[i] = s[i];
}

__global__ __launch_bounds__(256) void k_i64_to_f32(const long long* __restrict__ s,
                                                    float* __restrict__ d, long n) {
  long i = (long)blockIdx.x * 256 + threadIdx.x;
  if (i < n) d[i] = (float)s[i];
}

__global__ __launch_bounds__(256) void k_gather_i64_f32(const long long* __restrict__ s,
                                                        const int* __restrict__ sel,
                                                        float* __restrict__ d, int n) {
  int i = blockIdx.x * 256 + threadIdx.x;
  if (i < n) d[i] = (float)s[sel[i]];
}

__global__ void k_gather_rows_f32(const float* __restrict__ src, const int* __restrict__ sel,
                                  float* __restrict__ dst, int C) {
  int r = blockIdx.x; int j = sel[r];
  for (int c = threadIdx.x; c < C; c += blockDim.x)
    dst[(long)r * C + c] = src[(long)j * C + c];
}

__global__ void k_gather_rows_bf16(const float* __restrict__ src, const int* __restrict__ sel,
                                   __bf16* __restrict__ dst, int C) {
  int r = blockIdx.x; int j = sel[r];
  for (int c = threadIdx.x; c < C; c += blockDim.x)
    dst[(long)r * C + c] = (__bf16)src[(long)j * C + c];
}

// W [cin,cout] fp32 -> Wt [cout, ldk] bf16 (zero pad K to ldk, mult of 32)
__global__ __launch_bounds__(256) void k_transpose_w(const float* __restrict__ W,
                                                     __bf16* __restrict__ Wt,
                                                     int cin, int cout, int ldk) {
  long i = (long)blockIdx.x * 256 + threadIdx.x;
  long tot = (long)cout * ldk;
  if (i >= tot) return;
  int c  = (int)(i / ldk);
  int k2 = (int)(i % ldk);
  Wt[i] = (k2 < cin) ? (__bf16)W[(long)k2 * cout + c] : (__bf16)0.0f;
}

// ------------------------- WMMA GEMM ---------------------------------------
// C[n,cout] = act(A[n,lda](bf16) @ Wt[cout,ldb](bf16)^T + bias)
// grid.x = n/128 (8 waves x 16 rows), grid.y = cout/32 (2 N-subtiles / wave).
__global__ __launch_bounds__(256)
void k_wmma_gemm(const __bf16* __restrict__ A, int lda,
                 const __bf16* __restrict__ Bt, int ldb,
                 const float* __restrict__ bias,
                 float* __restrict__ C, int ldc,
                 int kTiles, int relu) {
  // 32 rows (cols of W), padded by 8 bf16 -> conflict-free ds_load_b128
  __shared__ __bf16 sB[32 * (288 + 8)];
  const int ldsb = ldb + 8;
  const long tileN0 = (long)blockIdx.y * 32;

  // ---- stage B macro-tile into LDS (once per block) ----
  {
    const int CH = ldb >> 3;          // 16B chunks per row
    const int total = 32 * CH;
    for (int c = threadIdx.x; c < total; c += 256) {
      const int row = c / CH, kc = c - row * CH;
      const __bf16* gp = Bt + (tileN0 + row) * (long)ldb + kc * 8;
      __bf16* lp = sB + row * ldsb + kc * 8;
#ifdef ASYNC_LDS
      __builtin_amdgcn_global_load_async_to_lds_b128(
          (__attribute__((address_space(1))) v4i_gcc*)(v4i_gcc*)gp,
          (__attribute__((address_space(3))) v4i_gcc*)(v4i_gcc*)lp, 0, 0);
#else
      *(bf16x8*)lp = *(const bf16x8*)gp;
#endif
    }
#ifdef ASYNC_LDS
#if __has_builtin(__builtin_amdgcn_s_wait_asynccnt)
    __builtin_amdgcn_s_wait_asynccnt(0);
#else
    asm volatile("s_wait_asynccnt 0" ::: "memory");
#endif
#endif
    __syncthreads();
  }

  const int lane = threadIdx.x & 31;
  const int wave = threadIdx.x >> 5;
  const int hsel = lane >> 4;     // 0: lanes 0-15, 1: lanes 16-31
  const int l16  = lane & 15;
  const long tileM = ((long)blockIdx.x * 8 + wave) * 16;

  // A-fragment per ISA 16-bit A layout: lanes 0-15 K{0..7,16..23}, 16-31 K{8..15,24..31}
  const __bf16* ap  = A + (tileM + l16) * (long)lda + hsel * 8;
  // B-fragment from LDS: lanes 0-15 K{0..7,16..23}, lanes 16-31 K{8..15,24..31}, N = lane&15
  const __bf16* b0p = sB + l16 * ldsb + hsel * 8;
  const __bf16* b1p = sB + (16 + l16) * ldsb + hsel * 8;

  f32x8 acc0 = {}, acc1 = {};
  bf16x8 a0 = *(const bf16x8*)(ap);
  bf16x8 a1 = *(const bf16x8*)(ap + 16);
  for (int kt = 0; kt < kTiles; ++kt) {
    bf16x8 na0 = a0, na1 = a1;
    if (kt + 1 < kTiles) {          // pipeline next A fragment ahead of WMMA
      na0 = *(const bf16x8*)(ap + (kt + 1) * 32);
      na1 = *(const bf16x8*)(ap + (kt + 1) * 32 + 16);
      __builtin_prefetch(ap + (kt + 2) * 32, 0, 1);
    }
    bf16x16 af = __builtin_shufflevector(a0, a1, 0,1,2,3,4,5,6,7,8,9,10,11,12,13,14,15);
    bf16x16 bf0 = __builtin_shufflevector(*(const bf16x8*)(b0p + kt * 32), *(const bf16x8*)(b0p + kt * 32 + 16), 0,1,2,3,4,5,6,7,8,9,10,11,12,13,14,15);
    bf16x16 bf1 = __builtin_shufflevector(*(const bf16x8*)(b1p + kt * 32), *(const bf16x8*)(b1p + kt * 32 + 16), 0,1,2,3,4,5,6,7,8,9,10,11,12,13,14,15);
    acc0 = __builtin_amdgcn_wmma_f32_16x16x32_bf16(false, af, false, bf0,
                                                   (short)0, acc0, false, false);
    acc1 = __builtin_amdgcn_wmma_f32_16x16x32_bf16(false, af, false, bf1,
                                                   (short)0, acc1, false, false);
    a0 = na0; a1 = na1;
  }
  const long col0 = tileN0 + l16;
  const float bv0 = bias ? bias[col0] : 0.0f;
  const float bv1 = bias ? bias[col0 + 16] : 0.0f;
#pragma unroll
  for (int r = 0; r < 8; ++r) {
    long row = tileM + hsel * 8 + r;   // lanes 0-15: M=r, lanes 16-31: M=r+8
    float v0 = acc0[r] + bv0;
    float v1 = acc1[r] + bv1;
    if (relu) { v0 = fmaxf(v0, 0.0f); v1 = fmaxf(v1, 0.0f); }
    C[row * (long)ldc + col0]      = v0;
    C[row * (long)ldc + col0 + 16] = v1;
  }
}

// ------------------------- LayerNorm (one wave / row) -----------------------
template <int PER>  // C = PER*32
__global__ __launch_bounds__(256)
void k_ln_rows(float* __restrict__ X, const float* __restrict__ g,
               const float* __restrict__ be, __bf16* __restrict__ Xbf, long n) {
  constexpr int C = PER * 32;
  const int lane = threadIdx.x & 31, wave = threadIdx.x >> 5;
  long row = (long)blockIdx.x * 8 + wave;
  if (row >= n) return;
  float* xr = X + row * C;
  float vals[PER];
  float s = 0.f;
#pragma unroll
  for (int t = 0; t < PER; ++t) { vals[t] = xr[lane + t * 32]; s += vals[t]; }
#pragma unroll
  for (int o = 16; o > 0; o >>= 1) s += __shfl_xor(s, o, 32);
  const float m = s * (1.0f / C);
  float v = 0.f;
#pragma unroll
  for (int t = 0; t < PER; ++t) { float d = vals[t] - m; v += d * d; }
#pragma unroll
  for (int o = 16; o > 0; o >>= 1) v += __shfl_xor(v, o, 32);
  const float rstd = rsqrtf(v * (1.0f / C) + 1e-5f);
#pragma unroll
  for (int t = 0; t < PER; ++t) {
    int c = lane + t * 32;
    float y = (vals[t] - m) * rstd * g[c] + be[c];
    xr[c] = y;
    if (Xbf) Xbf[row * C + c] = (__bf16)y;
  }
}

// ------------------------- FPS (single 32-wave workgroup) -------------------
__global__ __launch_bounds__(1024)
void k_fps(const float* __restrict__ pos, float* __restrict__ dbuf,
           int* __restrict__ sel, int n, int m) {
  __shared__ float sval[1024];
  __shared__ int   sidx[1024];
  __shared__ int   s_last;
  int tid = threadIdx.x;
  for (int j = tid; j < n; j += 1024) dbuf[j] = 1e10f;
  if (tid == 0) { sel[0] = 0; s_last = 0; }
  __syncthreads();
  for (int i = 1; i < m; ++i) {
    int last = s_last;
    float lx = pos[last * 3], ly = pos[last * 3 + 1], lz = pos[last * 3 + 2];
    float best = -1.f; int bidx = 0;
    for (int j = tid; j < n; j += 1024) {
      float dx = pos[j * 3] - lx, dy = pos[j * 3 + 1] - ly, dz = pos[j * 3 + 2] - lz;
      float nd = fminf(dbuf[j], dx * dx + dy * dy + dz * dz);
      dbuf[j] = nd;
      if (nd > best) { best = nd; bidx = j; }
    }
    sval[tid] = best; sidx[tid] = bidx;
    __syncthreads();
    for (int o = 512; o > 0; o >>= 1) {
      if (tid < o && sval[tid + o] > sval[tid]) { sval[tid] = sval[tid + o]; sidx[tid] = sidx[tid + o]; }
      __syncthreads();
    }
    if (tid == 0) { sel[i] = sidx[0]; s_last = sidx[0]; }
    __syncthreads();
  }
}

// ------------------------- kNN (register top-16, LDS tiles) -----------------
__global__ __launch_bounds__(256)
void k_knn(const float* __restrict__ pos, int n, int* __restrict__ nbr) {
  __shared__ float sp[256 * 3];
  int i = blockIdx.x * 256 + threadIdx.x;
  float px = pos[i * 3], py = pos[i * 3 + 1], pz = pos[i * 3 + 2];
  float bd[16]; int bi[16];
#pragma unroll
  for (int t = 0; t < 16; ++t) { bd[t] = 1e30f; bi[t] = 0; }
  for (int t0 = 0; t0 < n; t0 += 256) {
    __syncthreads();
    int j = t0 + threadIdx.x;
    sp[threadIdx.x * 3 + 0] = pos[j * 3 + 0];
    sp[threadIdx.x * 3 + 1] = pos[j * 3 + 1];
    sp[threadIdx.x * 3 + 2] = pos[j * 3 + 2];
    __syncthreads();
    for (int jj = 0; jj < 256; ++jj) {
      int jg = t0 + jj;
      float dx = px - sp[jj * 3], dy = py - sp[jj * 3 + 1], dz = pz - sp[jj * 3 + 2];
      float d2 = dx * dx + dy * dy + dz * dz;
      if (jg == i) d2 = 1e30f;
      if (d2 < bd[15]) {  // sorted ascending insertion (fully unrolled)
        float dc = d2; int ic = jg;
#pragma unroll
        for (int t = 0; t < 16; ++t) {
          if (dc < bd[t]) { float td = bd[t]; int ti = bi[t]; bd[t] = dc; bi[t] = ic; dc = td; ic = ti; }
        }
      }
    }
  }
#pragma unroll
  for (int t = 0; t < 16; ++t) nbr[(long)i * 16 + t] = bi[t];
}

// ------------------------- edge features ------------------------------------
// Ae[e, 0:256]=f_i-f_j, [256:259]=pos_i-pos_j, [259:288]=0 (bf16); dp saved fp32
__global__ __launch_bounds__(128)
void k_build_edges(const float* __restrict__ f, const float* __restrict__ pos,
                   const int* __restrict__ nbr, __bf16* __restrict__ Ae,
                   float* __restrict__ dp) {
  long e = blockIdx.x;
  int i = (int)(e >> 4);
  int j = nbr[e];
  const float* fi = f + (long)i * 256;
  const float* fj = f + (long)j * 256;
  __bf16* ae = Ae + e * 288;
  for (int c = threadIdx.x; c < 256; c += 128) ae[c] = (__bf16)(fi[c] - fj[c]);
  if (threadIdx.x < 3) {
    float d = pos[i * 3 + threadIdx.x] - pos[j * 3 + threadIdx.x];
    ae[256 + threadIdx.x] = (__bf16)d;
    dp[e * 3 + threadIdx.x] = d;
  }
  for (int c = 259 + threadIdx.x; c < 288; c += 128) ae[c] = (__bf16)0.0f;
}

// pe_pre = relu(dp @ Wpe + bpe), K=3 done directly in VALU
__global__ __launch_bounds__(256)
void k_pe(const float* __restrict__ dp, const float* __restrict__ Wpe,
          const float* __restrict__ bpe, float* __restrict__ pe) {
  long e = blockIdx.x; int c = threadIdx.x;
  float d0 = dp[e * 3], d1 = dp[e * 3 + 1], d2 = dp[e * 3 + 2];
  float v = d0 * Wpe[c] + d1 * Wpe[256 + c] + d2 * Wpe[512 + c] + bpe[c];
  pe[e * 256 + c] = fmaxf(v, 0.0f);
}

// ------------------------- attention + aggregation --------------------------
__global__ __launch_bounds__(256)
void k_attn_agg(const float* __restrict__ qn, const float* __restrict__ pe,
                const float* __restrict__ kk, const float* __restrict__ Wij,
                const float* __restrict__ res, float* __restrict__ out) {
  __shared__ float sred[8];
  __shared__ float sscore[16];
  const long i = blockIdx.x;
  const int c = threadIdx.x, lane = c & 31, wave = c >> 5;
  const float qv = qn[i * 256 + c];
  for (int e = 0; e < 16; ++e) {
    const long eo = (i * 16 + e) * 256;
    float p = (qv + pe[eo + c]) * kk[eo + c];
#pragma unroll
    for (int o = 16; o > 0; o >>= 1) p += __shfl_xor(p, o, 32);
    if (lane == 0) sred[wave] = p;
    __syncthreads();
    if (c == 0) {
      float s2 = 0.f;
#pragma unroll
      for (int w = 0; w < 8; ++w) s2 += sred[w];
      sscore[e] = s2 * 0.0625f;  // / sqrt(256)
    }
    __syncthreads();
  }
  float mx = -1e30f;
#pragma unroll
  for (int e = 0; e < 16; ++e) mx = fmaxf(mx, sscore[e]);
  float wsum = 0.f; float w[16];
#pragma unroll
  for (int e = 0; e < 16; ++e) { w[e] = __expf(sscore[e] - mx); wsum += w[e]; }
  const float inv = 1.0f / wsum;
  float sagg = 0.f;
#pragma unroll
  for (int e = 0; e < 16; ++e) sagg += (w[e] * inv) * Wij[(i * 16 + e) * 256 + c];
  out[i * 256 + c] = sagg + res[i * 256 + c];
}

// ------------------------- virtual node helpers -----------------------------
__global__ __launch_bounds__(256)
void k_col_mean(const float* __restrict__ X, float* __restrict__ mean, int n, int C) {
  __shared__ float sm[256];
  int c = blockIdx.x;
  float s = 0.f;
  for (int r = threadIdx.x; r < n; r += 256) s += X[(long)r * C + c];
  sm[threadIdx.x] = s; __syncthreads();
  for (int o = 128; o > 0; o >>= 1) {
    if (threadIdx.x < o) sm[threadIdx.x] += sm[threadIdx.x + o];
    __syncthreads();
  }
  if (threadIdx.x == 0) mean[c] = sm[0] / (float)n;
}

__global__ __launch_bounds__(256)
void k_vec_mm(const float* __restrict__ v, const float* __restrict__ W,
              const float* __restrict__ b, float* __restrict__ y, int d) {
  int c = threadIdx.x;
  if (c >= d) return;
  float s = b[c];
  for (int k2 = 0; k2 < d; ++k2) s += v[k2] * W[(long)k2 * d + c];
  y[c] = s;
}

__global__ __launch_bounds__(256)
void k_add_vec(float* __restrict__ X, const float* __restrict__ v, long tot, int C) {
  long i = (long)blockIdx.x * 256 + threadIdx.x;
  if (i < tot) X[i] += v[i % C];
}

// ---------------------------------------------------------------------------
extern "C" void kernel_launch(void* const* d_in, const int* in_sizes, int n_in,
                              void* d_out, int out_size, void* d_ws, size_t ws_size,
                              hipStream_t stream) {
  (void)in_sizes; (void)n_in; (void)out_size; (void)ws_size;
  int pi = 0;
  auto F = [&]() -> const float* { return (const float*)d_in[pi++]; };

  // params flattened in reference dict insertion order
  const float *s0W = F(), *s0b = F(), *s0g = F(), *s0be = F();
  const float *v0Wa = F(), *v0ba = F(), *v0Wd = F(), *v0bd = F(), *v0g = F(), *v0be = F();
  const float *l1Wf = F(), *l1bf = F(), *l1gf = F(), *l1bef = F(),
              *l1Wwf = F(), *l1bwf = F(), *l1gwf = F(), *l1bewf = F(),
              *l1Wq = F(), *l1bq = F(), *l1Wk = F(), *l1bk = F(),
              *l1Wpe = F(), *l1bpe = F(), *l1gpe = F(), *l1bepe = F(),
              *l1gfn = F(), *l1befn = F(), *l1Wr = F(), *l1br = F();
  const float *l2Wf = F(), *l2bf = F(), *l2gf = F(), *l2bef = F(),
              *l2Wwf = F(), *l2bwf = F(), *l2gwf = F(), *l2bewf = F(),
              *l2Wq = F(), *l2bq = F(), *l2Wk = F(), *l2bk = F(),
              *l2Wpe = F(), *l2bpe = F(), *l2gpe = F(), *l2bepe = F(),
              *l2gfn = F(), *l2befn = F();
  const float *v1Wa = F(), *v1ba = F(), *v1Wd = F(), *v1bd = F(), *v1g = F(), *v1be = F();
  const float* x   = F();
  const float* pos = F();
  const long long* labels = (const long long*)d_in[pi++];

  // ---- workspace layout ----
  char* base = (char*)d_ws;
  size_t off = 0;
  auto A8 = [&](size_t bytes) -> char* {
    char* p = base + off;
    off += (bytes + 255) & ~(size_t)255;
    return p;
  };
  __bf16* wt      = (__bf16*)A8((size_t)256 * 288 * 2);
  __bf16* xbf     = (__bf16*)A8((size_t)16384 * 64 * 2);
  float*  h       = (float*) A8((size_t)16384 * 128 * 4);
  float*  fpsd    = (float*) A8((size_t)16384 * 4);
  int*    sel     = (int*)   A8((size_t)8192 * 4);
  float*  pos1    = (float*) A8((size_t)8192 * 3 * 4);
  int*    nbr     = (int*)   A8((size_t)8192 * 16 * 4);
  __bf16* h1bf    = (__bf16*)A8((size_t)8192 * 128 * 2);
  float*  fbuf    = (float*) A8((size_t)8192 * 256 * 4);
  __bf16* fbf     = (__bf16*)A8((size_t)8192 * 256 * 2);
  __bf16* Aedge   = (__bf16*)A8((size_t)131072 * 288 * 2);
  float*  dp      = (float*) A8((size_t)131072 * 3 * 4);
  float*  Wij     = (float*) A8((size_t)131072 * 256 * 4);
  __bf16* Wijbf   = (__bf16*)A8((size_t)131072 * 256 * 2);
  float*  peb     = (float*) A8((size_t)131072 * 256 * 4);
  float*  kk      = (float*) A8((size_t)131072 * 256 * 4);
  float*  qn      = (float*) A8((size_t)8192 * 256 * 4);
  float*  resb    = (float*) A8((size_t)8192 * 256 * 4);
  float*  l1out   = (float*) A8((size_t)8192 * 256 * 4);
  __bf16* l1outbf = (__bf16*)A8((size_t)8192 * 256 * 2);
  float*  l2out   = (float*) A8((size_t)8192 * 256 * 4);
  float*  meanb   = (float*) A8(256 * 4);
  float*  gcb     = (float*) A8(256 * 4);
  float*  vvec    = (float*) A8(256 * 4);

  // ---- GEMM helper: transpose/convert W then WMMA GEMM ----
  auto gemm = [&](const __bf16* A_, int lda, const float* W_, int cin, int cout,
                  const float* bias, float* C_, long n, int relu) {
    int ldk = (cin + 31) & ~31;
    dim3 tg((unsigned)(((long)cout * ldk + 255) / 256));
    k_transpose_w<<<tg, 256, 0, stream>>>(W_, wt, cin, cout, ldk);
    dim3 g((unsigned)(n / 128), (unsigned)(cout / 32));
    k_wmma_gemm<<<g, 256, 0, stream>>>(A_, lda, wt, ldk, bias, C_, cout, ldk / 32, relu);
  };

  auto agt = [&](const __bf16* inBf, int cin, const float* resSrc,
                 const float* Wf, const float* bff, const float* gf, const float* bef,
                 const float* Wwf, const float* bwf, const float* gwf, const float* bewf,
                 const float* Wq, const float* bq, const float* Wk, const float* bk,
                 const float* Wpe, const float* bpe, const float* gpe, const float* bepe,
                 const float* gfn, const float* befn,
                 const float* Wr, const float* br,
                 float* outF, __bf16* outBf) {
    const long n = 8192, E = 131072;
    gemm(inBf, cin, Wf, cin, 256, bff, fbuf, n, 1);
    k_ln_rows<8><<<dim3((unsigned)(n / 8)), 256, 0, stream>>>(fbuf, gf, bef, fbf, n);
    k_build_edges<<<dim3((unsigned)E), 128, 0, stream>>>(fbuf, pos1, nbr, Aedge, dp);
    gemm(Aedge, 288, Wwf, 259, 256, bwf, Wij, E, 1);
    k_ln_rows<8><<<dim3((unsigned)(E / 8)), 256, 0, stream>>>(Wij, gwf, bewf, Wijbf, E);
    k_pe<<<dim3((unsigned)E), 256, 0, stream>>>(dp, Wpe, bpe, peb);
    k_ln_rows<8><<<dim3((unsigned)(E / 8)), 256, 0, stream>>>(peb, gpe, bepe, (__bf16*)nullptr, E);
    gemm(fbf, 256, Wq, 256, 256, bq, qn, n, 0);
    gemm(Wijbf, 256, Wk, 256, 256, bk, kk, E, 0);
    const float* resPtr = resSrc;
    if (Wr) { gemm(inBf, cin, Wr, cin, 256, br, resb, n, 0); resPtr = resb; }
    k_attn_agg<<<dim3((unsigned)n), 256, 0, stream>>>(qn, peb, kk, Wij, resPtr, outF);
    k_ln_rows<8><<<dim3((unsigned)(n / 8)), 256, 0, stream>>>(outF, gfn, befn, outBf, n);
  };

  auto vnode = [&](float* X, long n, int d,
                   const float* Wa, const float* ba, const float* Wd_, const float* bd_,
                   const float* g, const float* be) {
    k_col_mean<<<dim3((unsigned)d), 256, 0, stream>>>(X, meanb, (int)n, d);
    k_vec_mm<<<dim3(1), 256, 0, stream>>>(meanb, Wa, ba, gcb, d);
    if (d == 128) k_ln_rows<4><<<dim3(1), 256, 0, stream>>>(gcb, g, be, (__bf16*)nullptr, 1);
    else          k_ln_rows<8><<<dim3(1), 256, 0, stream>>>(gcb, g, be, (__bf16*)nullptr, 1);
    k_vec_mm<<<dim3(1), 256, 0, stream>>>(gcb, Wd_, bd_, vvec, d);
    long tot = n * d;
    k_add_vec<<<dim3((unsigned)((tot + 255) / 256)), 256, 0, stream>>>(X, vvec, tot, d);
  };

  // ---- stage 0: MLP + LN + virtual node ----
  k_f32_to_bf16<<<dim3((unsigned)((16384L * 64 + 255) / 256)), 256, 0, stream>>>(
      x, xbf, (long)16384 * 64);
  gemm(xbf, 64, s0W, 64, 128, s0b, h, 16384, 1);
  k_ln_rows<4><<<dim3(16384 / 8), 256, 0, stream>>>(h, s0g, s0be, (__bf16*)nullptr, 16384);
  vnode(h, 16384, 128, v0Wa, v0ba, v0Wd, v0bd, v0g, v0be);

  // ---- FPS downsample, gathers, kNN graph ----
  k_fps<<<dim3(1), 1024, 0, stream>>>(pos, fpsd, sel, 16384, 8192);
  k_gather_rows_f32<<<dim3(8192), 32, 0, stream>>>(pos, sel, pos1, 3);
  k_gather_rows_bf16<<<dim3(8192), 128, 0, stream>>>(h, sel, h1bf, 128);
  k_knn<<<dim3(8192 / 256), 256, 0, stream>>>(pos1, 8192, nbr);

  // ---- AGT layer 1 (128 -> 256, with Wr residual) ----
  agt(h1bf, 128, nullptr,
      l1Wf, l1bf, l1gf, l1bef, l1Wwf, l1bwf, l1gwf, l1bewf,
      l1Wq, l1bq, l1Wk, l1bk, l1Wpe, l1bpe, l1gpe, l1bepe,
      l1gfn, l1befn, l1Wr, l1br, l1out, l1outbf);

  // ---- AGT layer 2 (256 -> 256, identity residual) ----
  agt(l1outbf, 256, l1out,
      l2Wf, l2bf, l2gf, l2bef, l2Wwf, l2bwf, l2gwf, l2bewf,
      l2Wq, l2bq, l2Wk, l2bk, l2Wpe, l2bpe, l2gpe, l2bepe,
      l2gfn, l2befn, nullptr, nullptr, l2out, (__bf16*)nullptr);

  // ---- final virtual node ----
  vnode(l2out, 8192, 256, v1Wa, v1ba, v1Wd, v1bd, v1g, v1be);

  // ---- assemble outputs: feats(x,h,h1), poss(pos,pos,pos1), labs ----
  float* o = (float*)d_out;
  auto cpf = [&](const float* s, long cnt) {
    k_copy_f32<<<dim3((unsigned)((cnt + 255) / 256)), 256, 0, stream>>>(s, o, cnt);
    o += cnt;
  };
  cpf(x, (long)16384 * 64);
  cpf(h, (long)16384 * 128);
  cpf(l2out, (long)8192 * 256);
  cpf(pos, (long)16384 * 3);
  cpf(pos, (long)16384 * 3);
  cpf(pos1, (long)8192 * 3);
  k_i64_to_f32<<<dim3((16384 + 255) / 256), 256, 0, stream>>>(labels, o, 16384); o += 16384;
  k_i64_to_f32<<<dim3((16384 + 255) / 256), 256, 0, stream>>>(labels, o, 16384); o += 16384;
  k_gather_i64_f32<<<dim3((8192 + 255) / 256), 256, 0, stream>>>(labels, sel, o, 8192);
}
